// Attention_35974646071715
// MI455X (gfx1250) — matrix operand order repaired and checked by reading, not compiled, not run on hardware
//
#include <hip/hip_runtime.h>
#include <hip/hip_bf16.h>

// ---------------------------------------------------------------------------
// Multi-head attention, CDNA5 (gfx1250), wave32 WMMA f16 pipeline + TDM.
//   B=2, N=2048, C=1024, H=16, D=64
// ---------------------------------------------------------------------------

typedef __attribute__((ext_vector_type(16))) _Float16 v16h;
typedef __attribute__((ext_vector_type(8)))  float    v8f;
typedef __attribute__((ext_vector_type(4)))  unsigned int u32x4;
typedef __attribute__((ext_vector_type(4)))  int      i32x4;
typedef __attribute__((ext_vector_type(8)))  int      i32x8;

#define BATCH 2
#define SEQ   2048
#define CDIM  1024
#define HEADS 16
#define HDIM  64

// XOR butterfly via ds_swizzle (group-of-32, imm pattern: xor<<10 | and_mask 0x1f)
#define SWZ_XOR(x, m) \
    __int_as_float(__builtin_amdgcn_ds_swizzle(__float_as_int(x), (((m) << 10) | 0x1f)))

// ---- WMMA wrapper ----------------------------------------------------------
__device__ __forceinline__ v8f wmma16(v16h a, v16h b, v8f c) {
    return __builtin_amdgcn_wmma_f32_16x16x32_f16(
        false, a, false, b, (short)0, c, false, false);
}

// ---- Fragment loaders ------------------------------------------------------
// A-matrix 16x32 f16: lanes 0-15 row M=lane, K halves {0..15};
// lanes 16-31 row M=lane-16, K halves {16..31}.
__device__ __forceinline__ v16h load_a_f16(const _Float16* row, int kbase, int sel) {
    union { v16h v; uint4 q[2]; } u;
    u.q[0] = *reinterpret_cast<const uint4*>(row + kbase + sel * 16);
    u.q[1] = *reinterpret_cast<const uint4*>(row + kbase + sel * 16 + 8);
    return u.v;
}

// Same A layout, fused f32->f16 convert.
__device__ __forceinline__ v16h load_a_f32(const float* row, int kbase, int sel) {
    const float4* p0 = reinterpret_cast<const float4*>(row + kbase + sel * 16);
    const float4* p1 = reinterpret_cast<const float4*>(row + kbase + sel * 16 + 8);
    float4 f0 = p0[0], f1 = p0[1], f2 = p1[0], f3 = p1[1];
    v16h v;
    v[0]  = (_Float16)f0.x; v[1]  = (_Float16)f0.y; v[2]  = (_Float16)f0.z; v[3]  = (_Float16)f0.w;
    v[4]  = (_Float16)f1.x; v[5]  = (_Float16)f1.y; v[6]  = (_Float16)f1.z; v[7]  = (_Float16)f1.w;
    v[8]  = (_Float16)f2.x; v[9]  = (_Float16)f2.y; v[10] = (_Float16)f2.z; v[11] = (_Float16)f2.w;
    v[12] = (_Float16)f3.x; v[13] = (_Float16)f3.y; v[14] = (_Float16)f3.z; v[15] = (_Float16)f3.w;
    return v;
}

// B-matrix 32x16 f16: lane = column N%16, 16 contiguous K halves, K-base (lane/16)*16.
__device__ __forceinline__ v16h load_b_f16(const _Float16* p) {
    union { v16h v; uint4 q[2]; } u;
    u.q[0] = *reinterpret_cast<const uint4*>(p);
    u.q[1] = *reinterpret_cast<const uint4*>(p + 8);
    return u.v;
}

// ---- Tensor Data Mover: async 2D tile (f16) global -> LDS ------------------
// D# per ISA 8.3/8.4: group0 = {flags, lds_addr, global_addr, type=2},
// group1 = {data_size=2B, tensor_dim0/1, tile_dim0/1, dim0_stride}; groups 2/3 zero.
// This toolchain exposes the 6-arg builtin: (u32x4, i32x8, i32x4, i32x4, i32x8, cpol).
__device__ __forceinline__ void tdm_load_2d(unsigned lds_off, const void* gptr,
                                            unsigned tile0, unsigned tile1,
                                            unsigned long long stride_elems) {
    unsigned long long ga = (unsigned long long)(uintptr_t)gptr;
    u32x4 g0;
    g0[0] = 1u;                                              // count=1, user D#
    g0[1] = lds_off;                                         // LDS byte address
    g0[2] = (unsigned)ga;                                    // global_addr[31:0]
    g0[3] = (unsigned)((ga >> 32) & 0x01ffffffu) | (2u << 30); // addr[56:32] | type=2
    i32x8 g1;
    g1[0] = 0x00010000;                                      // data_size=1 -> 2 bytes
    g1[1] = (int)((tile0 & 0xffffu) << 16);                  // tensor_dim0[15:0]
    g1[2] = (int)(((tile0 >> 16) & 0xffffu) | ((tile1 & 0xffffu) << 16)); // td0 hi | td1 lo
    g1[3] = (int)(((tile1 >> 16) & 0xffffu) | ((tile0 & 0xffffu) << 16)); // td1 hi | tile_dim0
    g1[4] = (int)(tile1 & 0xffffu);                          // tile_dim1 (tile_dim2=0)
    g1[5] = (int)(stride_elems & 0xffffffffull);             // dim0_stride[31:0]
    g1[6] = (int)((stride_elems >> 32) & 0xffffull);         // dim0_stride[47:32]
    g1[7] = 0;
    i32x4 z4 = {0, 0, 0, 0};
    i32x8 z8 = {0, 0, 0, 0, 0, 0, 0, 0};
    __builtin_amdgcn_tensor_load_to_lds(g0, g1, z4, z4, z8, 0);
}

// ---------------------------------------------------------------------------
// Kernel 1: weight transpose + f32->f16.  W[K][N] fp32 -> WT[N][K] f16
// ---------------------------------------------------------------------------
__global__ __launch_bounds__(256) void transpose_cvt(const float* __restrict__ W,
                                                     _Float16* __restrict__ WT) {
    int idx = blockIdx.x * 256 + threadIdx.x;
    int k = idx >> 10;
    int n = idx & 1023;
    WT[n * CDIM + k] = (_Float16)W[idx];
}

// ---------------------------------------------------------------------------
// Kernel 2: WMMA GEMM  Y = (X @ W + bias) * ascale.  Wave tile 32x64.
// MODE 0/1: Q/K heads f16 (B,H,N,D)  MODE 2: V f16 transposed (B,H,D,N)
// MODE 3: fp32 (B,N,C)
// ---------------------------------------------------------------------------
template <int MODE>
__global__ __launch_bounds__(256) void wmma_gemm(const float* __restrict__ X,
                                                 const _Float16* __restrict__ WT,
                                                 const float* __restrict__ bias,
                                                 void* __restrict__ outp,
                                                 float ascale) {
    const int lane = threadIdx.x & 31;
    const int wv   = threadIdx.x >> 5;
    const int gw   = blockIdx.x * 8 + wv;     // 0..2047
    const int mt   = gw >> 4;                 // 32-row block
    const int nt   = gw & 15;                 // 64-col block
    const int sel  = lane >> 4;
    const int colq = lane & 15;

    const float* arow0 = X + (size_t)(mt * 32 + colq) * CDIM;
    const float* arow1 = arow0 + (size_t)16 * CDIM;
    const _Float16* brow[4];
#pragma unroll
    for (int c = 0; c < 4; ++c)
        brow[c] = WT + (size_t)(nt * 64 + c * 16 + colq) * CDIM;

    v8f acc[2][4] = {};

    for (int ks = 0; ks < CDIM; ks += 32) {
        __builtin_prefetch(arow0 + ks + 128, 0, 0);
        v16h a0 = load_a_f32(arow0, ks, sel);
        v16h a1 = load_a_f32(arow1, ks, sel);
#pragma unroll
        for (int c = 0; c < 4; ++c) {
            v16h b = load_b_f16(brow[c] + ks + sel * 16);
            acc[0][c] = wmma16(a0, b, acc[0][c]);
            acc[1][c] = wmma16(a1, b, acc[1][c]);
        }
    }

    float bv[4];
#pragma unroll
    for (int c = 0; c < 4; ++c) bv[c] = bias[nt * 64 + c * 16 + colq];

#pragma unroll
    for (int rt = 0; rt < 2; ++rt) {
#pragma unroll
        for (int c = 0; c < 4; ++c) {
#pragma unroll
            for (int j = 0; j < 8; ++j) {
                float val = (acc[rt][c][j] + bv[c]) * ascale;
                int m   = mt * 32 + rt * 16 + j + 8 * sel;
                int col = nt * 64 + c * 16 + colq;
                if constexpr (MODE == 3) {
                    ((float*)outp)[(size_t)m * CDIM + col] = val;
                } else {
                    int b = m >> 11, n = m & 2047;
                    int h = col >> 6, d = col & 63;
                    _Float16* o = (_Float16*)outp;
                    if constexpr (MODE == 2) {
                        o[(((size_t)b * HEADS + h) * HDIM + d) * SEQ + n] = (_Float16)val;
                    } else {
                        o[(((size_t)b * HEADS + h) * SEQ + n) * HDIM + d] = (_Float16)val;
                    }
                }
            }
        }
    }
}

// ---------------------------------------------------------------------------
// Kernel 3: flash attention.
//   Block = one (b,h), 8 waves = 8 query tiles of 16 rows.
//   TDM double-buffers the shared 64x64 K-tile and V-tile into LDS,
//   overlapped with WMMA; per-wave P transpose through LDS.
// ---------------------------------------------------------------------------
__global__ __launch_bounds__(256) void flash_attn(const _Float16* __restrict__ Qh,
                                                  const _Float16* __restrict__ Kh,
                                                  const _Float16* __restrict__ Vt,
                                                  float* __restrict__ attn) {
    __shared__ _Float16 kstage[2][64][64];   // 16 KB : keys x dims
    __shared__ _Float16 vstage[2][64][64];   // 16 KB : dims x keys
    __shared__ _Float16 plds[8][16][64];     // 16 KB : per-wave P transpose

    const int lane = threadIdx.x & 31;
    const int wv   = threadIdx.x >> 5;
    const int bh   = blockIdx.x >> 4;        // 0..31 : (b,h)
    const int qg   = blockIdx.x & 15;        // query group
    const int qt   = qg * 8 + wv;            // query tile 0..127
    const int b    = bh >> 4;
    const int h    = bh & 15;
    const int sel  = lane >> 4;
    const int colq = lane & 15;

    const _Float16* Qb = Qh + (size_t)bh * SEQ * HDIM;
    const _Float16* Kb = Kh + (size_t)bh * SEQ * HDIM;
    const _Float16* Vb = Vt + (size_t)bh * HDIM * SEQ;

    // Q fragments (16 rows x 64 dims), resident all kernel.
    const _Float16* qrow = Qb + (size_t)(qt * 16 + colq) * HDIM;
    v16h qa0 = load_a_f16(qrow, 0, sel);
    v16h qa1 = load_a_f16(qrow, 32, sel);

    float mi[8], li[8];
    v8f o[4] = {};
#pragma unroll
    for (int j = 0; j < 8; ++j) { mi[j] = -1e30f; li[j] = 0.0f; }

    // Prologue: async-stage first K/V tile.
    if (wv == 0) {
        tdm_load_2d((unsigned)(uintptr_t)&kstage[0][0][0], Kb, 64, 64, 64ull);
        tdm_load_2d((unsigned)(uintptr_t)&vstage[0][0][0], Vb, 64, 64, (unsigned long long)SEQ);
    }

    int buf = 0;
    for (int kt = 0; kt < SEQ; kt += 64) {
        const int ktn = kt + 64;
        __syncthreads();   // previous buffers fully consumed
        if (wv == 0) {
            if (ktn < SEQ) {
                tdm_load_2d((unsigned)(uintptr_t)&kstage[buf ^ 1][0][0],
                            Kb + (size_t)ktn * HDIM, 64, 64, 64ull);
                tdm_load_2d((unsigned)(uintptr_t)&vstage[buf ^ 1][0][0],
                            Vb + ktn, 64, 64, (unsigned long long)SEQ);
                __builtin_amdgcn_s_wait_tensorcnt(2);   // current tile done
            } else {
                __builtin_amdgcn_s_wait_tensorcnt(0);
            }
        }
        __syncthreads();   // current buffers ready for all waves

        // ---- S = Q * K^T : 4 score tiles of 16x16 over 64 keys ----
        v8f s[4];
#pragma unroll
        for (int st = 0; st < 4; ++st) {
            s[st] = (v8f){};
            const _Float16* kr = &kstage[buf][st * 16 + colq][0];
            s[st] = wmma16(qa0, load_b_f16(kr + sel * 16), s[st]);
            s[st] = wmma16(qa1, load_b_f16(kr + 32 + sel * 16), s[st]);
        }

        // ---- online softmax over 64 keys ----
        float p[4][8];
#pragma unroll
        for (int j = 0; j < 8; ++j) {
            float t = fmaxf(fmaxf(s[0][j], s[1][j]), fmaxf(s[2][j], s[3][j]));
            t = fmaxf(t, SWZ_XOR(t, 1));
            t = fmaxf(t, SWZ_XOR(t, 2));
            t = fmaxf(t, SWZ_XOR(t, 4));
            t = fmaxf(t, SWZ_XOR(t, 8));
            float mn    = fmaxf(mi[j], t);
            float alpha = __expf(mi[j] - mn);
            float rs = 0.0f;
#pragma unroll
            for (int st = 0; st < 4; ++st) {
                p[st][j] = __expf(s[st][j] - mn);
                rs += p[st][j];
            }
            rs += SWZ_XOR(rs, 1);
            rs += SWZ_XOR(rs, 2);
            rs += SWZ_XOR(rs, 4);
            rs += SWZ_XOR(rs, 8);
            li[j] = li[j] * alpha + rs;
            mi[j] = mn;
#pragma unroll
            for (int c = 0; c < 4; ++c) o[c][j] *= alpha;
        }

        // ---- P : C layout -> A layout through per-wave LDS ----
#pragma unroll
        for (int j = 0; j < 8; ++j) {
            int row = j + 8 * sel;
#pragma unroll
            for (int st = 0; st < 4; ++st)
                plds[wv][row][st * 16 + colq] = (_Float16)p[st][j];
        }
        asm volatile("s_wait_dscnt 0" ::: "memory");
        v16h pa0 = load_a_f16(&plds[wv][colq][0], 0, sel);
        v16h pa1 = load_a_f16(&plds[wv][colq][0], 32, sel);

        // ---- O += P * V : 4 output d-tiles, K = 64 keys ----
#pragma unroll
        for (int c = 0; c < 4; ++c) {
            const _Float16* vr = &vstage[buf][c * 16 + colq][0];
            o[c] = wmma16(pa0, load_b_f16(vr + sel * 16), o[c]);
            o[c] = wmma16(pa1, load_b_f16(vr + 32 + sel * 16), o[c]);
        }
        buf ^= 1;
    }

    // ---- normalize and write (B,N,C) fp32 ----
#pragma unroll
    for (int j = 0; j < 8; ++j) {
        float inv = 1.0f / li[j];
        int n = qt * 16 + j + 8 * sel;
        float* orow = attn + ((size_t)b * SEQ + n) * CDIM + h * HDIM;
#pragma unroll
        for (int c = 0; c < 4; ++c) orow[c * 16 + colq] = o[c][j] * inv;
    }
}

// ---------------------------------------------------------------------------
// Host launcher
// ---------------------------------------------------------------------------
extern "C" void kernel_launch(void* const* d_in, const int* in_sizes, int n_in,
                              void* d_out, int out_size, void* d_ws, size_t ws_size,
                              hipStream_t stream) {
    (void)in_sizes; (void)n_in; (void)out_size; (void)ws_size;
    const float* q  = (const float*)d_in[0];
    const float* k  = (const float*)d_in[1];
    const float* v  = (const float*)d_in[2];
    const float* Wq = (const float*)d_in[3];
    const float* bq = (const float*)d_in[4];
    const float* Wk = (const float*)d_in[5];
    const float* bk = (const float*)d_in[6];
    const float* Wv = (const float*)d_in[7];
    const float* bv = (const float*)d_in[8];
    const float* Wo = (const float*)d_in[9];
    const float* bo = (const float*)d_in[10];

    char* ws = (char*)d_ws;
    const size_t MB = 1024 * 1024;
    _Float16* WqT  = (_Float16*)(ws + 0 * MB);
    _Float16* WkT  = (_Float16*)(ws + 2 * MB);
    _Float16* WvT  = (_Float16*)(ws + 4 * MB);
    _Float16* WoT  = (_Float16*)(ws + 6 * MB);
    _Float16* Qh   = (_Float16*)(ws + 8 * MB);   // f16 (B,H,N,D)
    _Float16* KhT  = (_Float16*)(ws + 16 * MB);  // f16 (B,H,N,D)
    _Float16* VtT  = (_Float16*)(ws + 24 * MB);  // f16 (B,H,D,N)
    float*    attn = (float*)   (ws + 32 * MB);  // fp32 (B,N,C)

    transpose_cvt<<<4096, 256, 0, stream>>>(Wq, WqT);
    transpose_cvt<<<4096, 256, 0, stream>>>(Wk, WkT);
    transpose_cvt<<<4096, 256, 0, stream>>>(Wv, WvT);
    transpose_cvt<<<4096, 256, 0, stream>>>(Wo, WoT);

    wmma_gemm<0><<<256, 256, 0, stream>>>(q, WqT, bq, (void*)Qh, 0.125f);
    wmma_gemm<1><<<256, 256, 0, stream>>>(k, WkT, bk, (void*)KhT, 1.0f);
    wmma_gemm<2><<<256, 256, 0, stream>>>(v, WvT, bv, (void*)VtT, 1.0f);

    flash_attn<<<512, 256, 0, stream>>>(Qh, KhT, VtT, attn);

    wmma_gemm<3><<<256, 256, 0, stream>>>(attn, WoT, bo, d_out, 1.0f);
}
